// KANLayer_64355789963718
// MI455X (gfx1250) — compile-verified, matrix operand order checked by reading, not run on hardware
//
#include <hip/hip_runtime.h>

// ---------------------------------------------------------------------------
// KAN layer as a sparse-A GEMM on the CDNA5 WMMA pipe.
//   out[8192,256] = W[8192,2944] x C[256,2944]^T,  K = 128*23 = 2944 = 92*32
// W has exactly 2 adjacent nonzeros per 23-wide segment (linear B-spline taps)
// and is scatter-built in LDS per 32-row batch tile. C = coeffs flattened,
// converted once to f16 in workspace (L2-resident, reused by every WG).
// Each wave computes a 2x2 block of 16x16 output tiles: 4 WMMAs per K-chunk
// against 4 ds_load_b128 + 4 global_load_b128 (every fragment reused twice).
// ---------------------------------------------------------------------------

#define BATCH   8192
#define IN_F    128
#define OUT_F   256
#define GRID_SZ 20
#define LCOEFF  23                   // grid_size + order
#define KDIM    (IN_F * LCOEFF)      // 2944 = 92 * 32  (exact multiple of 32)
#define KCHUNKS (KDIM / 32)          // 92
#define BTILE   32                   // batch rows per workgroup (2 M-tiles)

typedef _Float16 v16h __attribute__((ext_vector_type(16)));
typedef _Float16 v8h  __attribute__((ext_vector_type(8)));
typedef float    v8f  __attribute__((ext_vector_type(8)));

// --- coeffs f32 -> f16 (natural [O, I*L] layout == B-matrix row-major) -----
__global__ void kan_convert_coeffs(const float* __restrict__ src,
                                   _Float16* __restrict__ dst, int n) {
  int i = blockIdx.x * blockDim.x + threadIdx.x;
  if (i < n) dst[i] = (_Float16)src[i];
}

static __device__ __forceinline__ v16h cat8(v8h lo, v8h hi) {
  return __builtin_shufflevector(lo, hi,
           0,1,2,3,4,5,6,7,8,9,10,11,12,13,14,15);
}

// --- main kernel: 256 threads = 8 waves (wave32), one 32-row B tile --------
__global__ void __launch_bounds__(256)
kan_wmma_kernel(const float* __restrict__ x,
                const _Float16* __restrict__ C16,
                float* __restrict__ out) {
  // 32 rows x 2944 cols of f16 sparse spline weights: 188,416 bytes of LDS.
  __shared__ __align__(16) _Float16 Wlds[BTILE * KDIM];

  const int tid   = threadIdx.x;
  const int wave  = tid >> 5;
  const int lane  = tid & 31;
  const int bTile = blockIdx.x;

  // ---- Phase 0: zero the W tile (46 x b128 stores per thread) ----
  {
    uint4 z; z.x = 0u; z.y = 0u; z.z = 0u; z.w = 0u;
    uint4* wp = (uint4*)Wlds;
    #pragma unroll
    for (int i = 0; i < (BTILE * KDIM * 2 / 16) / 256; ++i)   // 11776/256 = 46
      wp[tid + i * 256] = z;
  }
  __syncthreads();

  // ---- Phase 1: scatter-build W (2 ds_store_b16 per (b,j) pair) ----
  #pragma unroll
  for (int t = 0; t < (BTILE * IN_F) / 256; ++t) {            // 16 pairs/thread
    const int pair = tid + t * 256;
    const int b = pair >> 7;                                  // pair / 128
    const int j = pair & 127;                                 // pair % 128
    const float xv = x[(bTile * BTILE + b) * IN_F + j];
    const float s  = 1.0f / (1.0f + __expf(-xv));             // sigmoid
    const float fi = s * (float)(GRID_SZ - 1);                // in [0, 19)
    const int   k  = (int)fi;                                 // floor, k<=19
    const float w1 = fi - (float)k;
    const int base = b * KDIM + j * LCOEFF + k;               // k+1 <= 22 < 23
    Wlds[base]     = (_Float16)(1.0f - w1);
    Wlds[base + 1] = (_Float16)w1;
  }
  __syncthreads();

  // ---- Phase 2: K-loop of v_wmma_f32_16x16x32_f16, 2x2 tile block/wave ----
  // 16-bit fragment striping (ISA 7.12.2): lanes 0-15 take K [c, c+8) and
  // [c+16, c+24) with c=0; lanes 16-31 the same with c=8.
  const int lane15 = lane & 15;
  const int khalf  = (lane >> 4) << 3;                        // 0 or 8
  const int n0 = wave * 2;                                    // 2 N-tiles/wave
  const int n1 = n0 + 1;

  const _Float16* a0_base = &Wlds[(     lane15) * KDIM + khalf];  // M-tile 0
  const _Float16* a1_base = &Wlds[(16 + lane15) * KDIM + khalf];  // M-tile 1
  const _Float16* b0_base = C16 + (n0 * 16 + lane15) * KDIM + khalf;
  const _Float16* b1_base = C16 + (n1 * 16 + lane15) * KDIM + khalf;

  __builtin_prefetch(b0_base, 0, 0);                          // global_prefetch
  __builtin_prefetch(b1_base, 0, 0);

  v8f acc00 = {}; v8f acc01 = {};                             // (m0,n0) (m0,n1)
  v8f acc10 = {}; v8f acc11 = {};                             // (m1,n0) (m1,n1)
  for (int kc = 0; kc < KCHUNKS; ++kc) {
    const int off = kc * 32;

    v16h a0 = cat8(*(const v8h*)(a0_base + off),              // ds_load_b128 x2
                   *(const v8h*)(a0_base + off + 16));
    v16h a1 = cat8(*(const v8h*)(a1_base + off),
                   *(const v8h*)(a1_base + off + 16));
    v16h b0 = cat8(*(const v8h*)(b0_base + off),              // global_load_b128 x2
                   *(const v8h*)(b0_base + off + 16));
    v16h b1 = cat8(*(const v8h*)(b1_base + off),
                   *(const v8h*)(b1_base + off + 16));

    acc00 = __builtin_amdgcn_wmma_f32_16x16x32_f16(
               false, a0, false, b0, (short)0, acc00, false, false);
    acc01 = __builtin_amdgcn_wmma_f32_16x16x32_f16(
               false, a0, false, b1, (short)0, acc01, false, false);
    acc10 = __builtin_amdgcn_wmma_f32_16x16x32_f16(
               false, a1, false, b0, (short)0, acc10, false, false);
    acc11 = __builtin_amdgcn_wmma_f32_16x16x32_f16(
               false, a1, false, b1, (short)0, acc11, false, false);
  }

  // ---- Phase 3: store D (f32 16x16: VGPR r -> row r (+8 for lanes>=16)) ----
  const int mrow = (lane >> 4) * 8;
  float* o00 = out + (bTile * BTILE +      mrow) * OUT_F + n0 * 16 + lane15;
  float* o01 = out + (bTile * BTILE +      mrow) * OUT_F + n1 * 16 + lane15;
  float* o10 = out + (bTile * BTILE + 16 + mrow) * OUT_F + n0 * 16 + lane15;
  float* o11 = out + (bTile * BTILE + 16 + mrow) * OUT_F + n1 * 16 + lane15;
  #pragma unroll
  for (int r = 0; r < 8; ++r) {
    o00[r * OUT_F] = acc00[r];
    o01[r * OUT_F] = acc01[r];
    o10[r * OUT_F] = acc10[r];
    o11[r * OUT_F] = acc11[r];
  }
}

extern "C" void kernel_launch(void* const* d_in, const int* in_sizes, int n_in,
                              void* d_out, int out_size, void* d_ws, size_t ws_size,
                              hipStream_t stream) {
  const float* x      = (const float*)d_in[0];   // [8192, 128] f32
  const float* coeffs = (const float*)d_in[1];   // [256, 128, 23] f32
  float*       out    = (float*)d_out;           // [8192, 256] f32
  _Float16*    C16    = (_Float16*)d_ws;         // 753,664 f16 = 1.47 MB

  const int n = OUT_F * KDIM;                    // 753,664
  kan_convert_coeffs<<<(n + 255) / 256, 256, 0, stream>>>(coeffs, C16, n);
  kan_wmma_kernel<<<BATCH / BTILE, 256, 0, stream>>>(x, C16, out);
}
